// LSTM_GCN_78005196030237
// MI455X (gfx1250) — matrix operand order repaired and checked by reading, not compiled, not run on hardware
//
#include <hip/hip_runtime.h>
#include <hip/hip_bf16.h>
#include <cstddef>

// ---------------------------------------------------------------------------
// Types for CDNA5 WMMA / TDM
// ---------------------------------------------------------------------------
typedef __attribute__((ext_vector_type(16))) __bf16   v16bf;
typedef __attribute__((ext_vector_type(8)))  float    v8f;
typedef __attribute__((ext_vector_type(4))) unsigned  tdm_u4;
typedef __attribute__((ext_vector_type(8)))  int      tdm_i8;
typedef __attribute__((ext_vector_type(4)))  int      tdm_i4;

#if defined(__has_builtin)
#if __has_builtin(__builtin_amdgcn_tensor_load_to_lds)
#define HAVE_TDM 1
#endif
#endif

// ---------------------------------------------------------------------------
// Workspace layout (bytes)
// ---------------------------------------------------------------------------
static const size_t OFF_LSTMW = 0;                 // 256 KB  bf16 fragments (w_ih | w_hh)
static const size_t OFF_G1    = 262144;            // 32 KB   bf16 fragments gw1
static const size_t OFF_G2    = 294912;            // 32 KB   bf16 fragments gw2
static const size_t OFF_BC    = 327680;            // 2 KB    combined LSTM bias (f32)
static const size_t OFF_HN    = 524288;            // N*128 bf16 (hn; later reused for relu(x))
static const size_t OFF_XW    = (size_t)16 << 20;  // N*128 f32 (x @ W, both layers)
static const size_t OFF_ACC   = (size_t)48 << 20;  // N*128 f32 (layer-1 scatter accumulator)

// ---------------------------------------------------------------------------
// Small device helpers
// ---------------------------------------------------------------------------
__device__ __forceinline__ unsigned lds_off_of(const void* p) {
    // low 32 bits of the shared-aperture flat address == LDS byte offset
    return (unsigned)(unsigned long long)p;
}

__device__ __forceinline__ unsigned short f2bf(float f) {
    __bf16 b = (__bf16)f;
    return __builtin_bit_cast(unsigned short, b);
}

__device__ __forceinline__ float fsig(float x) {   // sigmoid via v_exp/v_rcp
    return __builtin_amdgcn_rcpf(1.0f + __expf(-x));
}
__device__ __forceinline__ float ftanh(float x) {  // tanh(x) = 1 - 2/(e^{2x}+1)
    return 1.0f - 2.0f * __builtin_amdgcn_rcpf(1.0f + __expf(2.0f * x));
}

__device__ __forceinline__ v8f wmma_bf16(v16bf a, v16bf b, v8f c) {
    return __builtin_amdgcn_wmma_f32_16x16x32_bf16(
        /*neg_a=*/false, a, /*neg_b=*/false, b,
        /*c_mod=*/(short)0, c, /*reuse_a=*/false, /*reuse_b=*/false);
}

__device__ __forceinline__ v16bf ldB(const unsigned short* p) {   // one 512-bf16 fragment, 32B/lane
    return *(const v16bf*)p;
}

// A-fragment (16x32 bf16, ISA layout) from a row-major f32 row pointer.
// lanes 0-15: M=lane,   K-runs at kt*32+0  and kt*32+16
// lanes16-31: M=lane-16,K-runs at kt*32+8  and kt*32+24
__device__ __forceinline__ v16bf a_frag_from_f32(const float* __restrict__ row, int kt, int lane) {
    const int k0 = kt * 32 + ((lane & 16) ? 8 : 0);
    const float4* p = (const float4*)(row + k0);
    float4 f0 = p[0], f1 = p[1];
    const float4* q = (const float4*)(row + k0 + 16);
    float4 f2 = q[0], f3 = q[1];
    v16bf r;
    r[0]=(__bf16)f0.x;  r[1]=(__bf16)f0.y;  r[2]=(__bf16)f0.z;  r[3]=(__bf16)f0.w;
    r[4]=(__bf16)f1.x;  r[5]=(__bf16)f1.y;  r[6]=(__bf16)f1.z;  r[7]=(__bf16)f1.w;
    r[8]=(__bf16)f2.x;  r[9]=(__bf16)f2.y;  r[10]=(__bf16)f2.z; r[11]=(__bf16)f2.w;
    r[12]=(__bf16)f3.x; r[13]=(__bf16)f3.y; r[14]=(__bf16)f3.z; r[15]=(__bf16)f3.w;
    return r;
}

// Same A-fragment from a row-major bf16 row pointer (global or LDS).
__device__ __forceinline__ v16bf a_frag_from_bf16(const unsigned short* __restrict__ row, int kt, int lane) {
    const int k0 = kt * 32 + ((lane & 16) ? 8 : 0);
    union { uint4 u[2]; v16bf v; } u;
    u.u[0] = *(const uint4*)(row + k0);
    u.u[1] = *(const uint4*)(row + k0 + 16);
    return u.v;
}

// ---------------------------------------------------------------------------
// Tensor Data Mover: 1-D bulk copy global -> LDS (8-byte elements)
// ---------------------------------------------------------------------------
#if HAVE_TDM
__device__ __forceinline__ void tdm_load_1d(const void* gsrc, unsigned lds_byte_off, unsigned nbytes) {
    const unsigned n8 = nbytes >> 3;                 // #8-byte elements (<= 65535)
    const unsigned long long ga = (unsigned long long)gsrc;
    tdm_u4 g0;
    g0[0] = 1u;                                      // count=1, user descriptor
    g0[1] = lds_byte_off;                            // lds_addr
    g0[2] = (unsigned)ga;                            // global_addr[31:0]
    g0[3] = (unsigned)((ga >> 32) & 0x01FFFFFFu) | (2u << 30); // addr[56:32], type=2 (image)
    tdm_i8 g1;
    g1[0] = (int)(3u << 16);                         // data_size=3 (8B), no multicast/pad
    g1[1] = (int)((n8 & 0xFFFFu) << 16);             // tensor_dim0[15:0]  (bits 63:48)
    g1[2] = (int)((n8 >> 16) & 0xFFFFu);             // tensor_dim0[31:16] (bits 79:64)
    g1[3] = (int)((n8 & 0xFFFFu) << 16);             // tile_dim0          (bits 127:112)
    g1[4] = 0;                                       // tile_dim1/2 unused
    g1[5] = (int)n8;                                 // tensor_dim0_stride[31:0]
    g1[6] = 0;
    g1[7] = 0;
    tdm_i4 z4 = {0, 0, 0, 0};
    tdm_i8 z8 = {0, 0, 0, 0, 0, 0, 0, 0};
    __builtin_amdgcn_tensor_load_to_lds(g0, g1, z4, z4, z8, 0);
}
#endif

// Stage nbytes from global (bf16-packed) into LDS; ends with a block barrier.
__device__ __forceinline__ void stage_to_lds(const unsigned short* g, unsigned short* l,
                                             int nbytes, int tid, int nthreads) {
#if HAVE_TDM
    if (tid < 32) {                                  // wave 0 issues one TDM descriptor
        tdm_load_1d(g, lds_off_of(l), (unsigned)nbytes);
        __builtin_amdgcn_s_wait_tensorcnt(0);
    }
#else
    const int n16 = nbytes >> 4;
    for (int i = tid; i < n16; i += nthreads)
        ((uint4*)l)[i] = ((const uint4*)g)[i];
#endif
    __syncthreads();
}

// ---------------------------------------------------------------------------
// Weight packing kernels: f32 row-major -> bf16 WMMA B-fragment layout.
// Fragment (ntile j, ktile kt) = 512 bf16, per-lane-contiguous (16 each):
//   lane<16 : N = 16j+lane,    K = 32kt + {0..7, 16..23}   (t = 0..15)
//   lane>=16: N = 16j+lane-16, K = 32kt + {8..15, 24..31}
// ---------------------------------------------------------------------------
__global__ __launch_bounds__(256) void k_pack_lstm(const float* __restrict__ wih,
                                                   const float* __restrict__ whh,
                                                   unsigned short* __restrict__ dst) {
    const int idx  = blockIdx.x * 256 + threadIdx.x;     // 0 .. 262143
    const int t    = idx & 15;
    const int lane = (idx >> 4) & 31;
    const int kt   = (idx >> 9) & 3;
    const int j    = (idx >> 11) & 31;
    const int mat  = idx >> 16;
    const int n = 16 * j + (lane & 15);
    const int k = 32 * kt + ((t < 8) ? 8 * (lane >> 4) + t : 16 + 8 * (lane >> 4) + (t - 8));
    const float* w = mat ? whh : wih;                    // (512,128): B = W^T -> B[k][n]=w[n][k]
    dst[idx] = f2bf(w[n * 128 + k]);
}

__global__ __launch_bounds__(256) void k_pack_gw(const float* __restrict__ g,
                                                 unsigned short* __restrict__ dst) {
    const int idx  = blockIdx.x * 256 + threadIdx.x;     // 0 .. 16383
    const int t    = idx & 15;
    const int lane = (idx >> 4) & 31;
    const int kt   = (idx >> 9) & 3;
    const int j    = (idx >> 11) & 7;
    const int n = 16 * j + (lane & 15);
    const int k = 32 * kt + ((t < 8) ? 8 * (lane >> 4) + t : 16 + 8 * (lane >> 4) + (t - 8));
    dst[idx] = f2bf(g[k * 128 + n]);                     // x@W: B[k][n]=gw[k][n]
}

__global__ __launch_bounds__(256) void k_bias_comb(const float* __restrict__ bih,
                                                   const float* __restrict__ bhh,
                                                   float* __restrict__ b) {
    const int i = blockIdx.x * 256 + threadIdx.x;
    if (i < 512) b[i] = bih[i] + bhh[i];
}

// ---------------------------------------------------------------------------
// LSTM: one block = 128 nodes, 8 waves x 16 rows. Weights (256 KB bf16) TDM'd
// to LDS once; h kept in LDS (bf16), c in VGPRs, for all 8 timesteps.
// Dynamic LDS: 256 KB weights + 32 KB hbuf = 288 KB (< 320 KB / WGP).
// ---------------------------------------------------------------------------
__global__ __launch_bounds__(256) void k_lstm(const float* __restrict__ x,       // (8,N,128)
                                              const unsigned short* __restrict__ wpack,
                                              const float* __restrict__ bc,      // (512,)
                                              unsigned short* __restrict__ hn,   // (N,128) bf16
                                              int N) {
    extern __shared__ char smem[];
    unsigned short* wlds = (unsigned short*)smem;              // [2][32][4][512] bf16
    unsigned short* hbuf = (unsigned short*)(smem + 262144);   // 128 x 128 bf16

    const int tid = threadIdx.x, wave = tid >> 5, lane = tid & 31;

    // h0 = 0
    for (int i = tid; i < 2048; i += 256) ((uint4*)hbuf)[i] = make_uint4(0u, 0u, 0u, 0u);
    stage_to_lds(wpack, wlds, 262144, tid, 256);

    const int rowbase = wave * 16;
    const int arow    = rowbase + (lane & 15);                 // local A-frag row
    int gnode = blockIdx.x * 128 + arow;
    if (gnode > N - 1) gnode = N - 1;                          // clamp tail (EXEC stays full)
    const int crow0 = rowbase + ((lane & 16) ? 8 : 0);         // local C first row
    const int ccol  = lane & 15;

    v8f cst[8];
#pragma unroll
    for (int jj = 0; jj < 8; ++jj) cst[jj] = (v8f){0, 0, 0, 0, 0, 0, 0, 0};

    for (int t = 0; t < 8; ++t) {
        const float* xrow = x + ((size_t)t * (size_t)N + (size_t)gnode) * 128;
        v16bf xa[4], ha[4];
#pragma unroll
        for (int kt = 0; kt < 4; ++kt) xa[kt] = a_frag_from_f32(xrow, kt, lane);
        const unsigned short* hrow = hbuf + arow * 128;
#pragma unroll
        for (int kt = 0; kt < 4; ++kt) ha[kt] = a_frag_from_bf16(hrow, kt, lane);

#pragma unroll
        for (int jj = 0; jj < 8; ++jj) {                       // hidden-column tile
            v8f acc[4];
#pragma unroll
            for (int g4 = 0; g4 < 4; ++g4) {                   // i, f, g, o
                const int j = jj + 8 * g4;                     // gate n-tile
                const float bv = bc[16 * j + ccol];
                const unsigned short* bw = wlds + (j * 4) * 512 + lane * 16;          // x-path frags
                const unsigned short* bh = wlds + 65536 + (j * 4) * 512 + lane * 16;  // h-path frags
                v8f a = (v8f){bv, bv, bv, bv, bv, bv, bv, bv};
                // depth-2 rotating prefetch of the 8 B-fragments: the loads for
                // fragment k+2 issue before the WMMA consuming fragment k+1, so the
                // DS waits become partial (dscnt<=2) instead of full stalls.
                v16bf b0 = ldB(bw + 0 * 512);
                v16bf b1 = ldB(bw + 1 * 512);
                a = wmma_bf16(xa[0], b0, a);  b0 = ldB(bw + 2 * 512);
                a = wmma_bf16(xa[1], b1, a);  b1 = ldB(bw + 3 * 512);
                a = wmma_bf16(xa[2], b0, a);  b0 = ldB(bh + 0 * 512);
                a = wmma_bf16(xa[3], b1, a);  b1 = ldB(bh + 1 * 512);
                a = wmma_bf16(ha[0], b0, a);  b0 = ldB(bh + 2 * 512);
                a = wmma_bf16(ha[1], b1, a);  b1 = ldB(bh + 3 * 512);
                a = wmma_bf16(ha[2], b0, a);
                a = wmma_bf16(ha[3], b1, a);
                acc[g4] = a;
            }
            v8f c = cst[jj];
            unsigned short hb[8];
#pragma unroll
            for (int r = 0; r < 8; ++r) {
                const float ig = fsig(acc[0][r]);
                const float fg = fsig(acc[1][r]);
                const float gg = ftanh(acc[2][r]);
                const float og = fsig(acc[3][r]);
                const float cc = fg * c[r] + ig * gg;
                c[r]  = cc;
                hb[r] = f2bf(og * ftanh(cc));
            }
            cst[jj] = c;
#pragma unroll
            for (int r = 0; r < 8; ++r)                        // wave-private rows: no barrier
                hbuf[(crow0 + r) * 128 + (16 * jj + ccol)] = hb[r];
        }
    }
    __syncthreads();
    // final hidden state -> global (bf16, coalesced 16B stores)
    for (int i = tid; i < 2048; i += 256) {
        const int lrow = i >> 4;
        const int node = blockIdx.x * 128 + lrow;
        if (node < N)
            ((uint4*)(hn + (size_t)node * 128))[i & 15] = ((const uint4*)hbuf)[i];
    }
}

// ---------------------------------------------------------------------------
// GCN GEMM: C(N,128) f32 = A(N,128) bf16 @ Bpack(128,128). B TDM'd to LDS.
// ---------------------------------------------------------------------------
__global__ __launch_bounds__(256) void k_gcn_gemm(const unsigned short* __restrict__ A,
                                                  const unsigned short* __restrict__ Bpack,
                                                  float* __restrict__ C, int N) {
    __shared__ unsigned short blds[16384];                     // 32 KB
    const int tid = threadIdx.x, wave = tid >> 5, lane = tid & 31;
    stage_to_lds(Bpack, blds, 32768, tid, 256);

    const int rowbase = wave * 16;
    int arow = blockIdx.x * 128 + rowbase + (lane & 15);
    if (arow > N - 1) arow = N - 1;
    const unsigned short* rp = A + (size_t)arow * 128;
    v16bf a[4];
#pragma unroll
    for (int kt = 0; kt < 4; ++kt) a[kt] = a_frag_from_bf16(rp, kt, lane);

    const int crow0 = blockIdx.x * 128 + rowbase + ((lane & 16) ? 8 : 0);
    const int ccol  = lane & 15;
#pragma unroll
    for (int j = 0; j < 8; ++j) {
        const unsigned short* bb = blds + (j * 4) * 512 + lane * 16;
        v8f acc = (v8f){0, 0, 0, 0, 0, 0, 0, 0};
        v16bf b0 = ldB(bb + 0 * 512);
        v16bf b1 = ldB(bb + 1 * 512);
        acc = wmma_bf16(a[0], b0, acc);  b0 = ldB(bb + 2 * 512);
        acc = wmma_bf16(a[1], b1, acc);  b1 = ldB(bb + 3 * 512);
        acc = wmma_bf16(a[2], b0, acc);
        acc = wmma_bf16(a[3], b1, acc);
#pragma unroll
        for (int r = 0; r < 8; ++r) {
            const int node = crow0 + r;
            if (node < N) C[(size_t)node * 128 + 16 * j + ccol] = acc[r];
        }
    }
}

// ---------------------------------------------------------------------------
// Elementwise / graph kernels
// ---------------------------------------------------------------------------
__global__ __launch_bounds__(256) void k_zero4(float* __restrict__ p, long long n4) {
    const long long i = (long long)blockIdx.x * 256 + threadIdx.x;
    if (i < n4) ((float4*)p)[i] = make_float4(0.f, 0.f, 0.f, 0.f);
}

// One thread per (edge, 4 columns): out[dst] += msg[src] with L2 f32 atomics.
__global__ __launch_bounds__(256) void k_scatter(const long long* __restrict__ edges,  // [2][E]
                                                 const float* __restrict__ msg,
                                                 float* __restrict__ out, long long E) {
    const long long idx = (long long)blockIdx.x * 256 + threadIdx.x;
    if (idx >= E * 32) return;
    const long long e = idx >> 5;
    const int       q = (int)(idx & 31);
    const long long s = edges[e];
    const long long d = edges[E + e];
    const float4 v = *(const float4*)(msg + s * 128 + q * 4);
    float* o = out + d * 128 + (long long)q * 4;
    atomicAdd(o + 0, v.x);
    atomicAdd(o + 1, v.y);
    atomicAdd(o + 2, v.z);
    atomicAdd(o + 3, v.w);
}

__global__ __launch_bounds__(256) void k_bias_relu_bf16(const float* __restrict__ in,
                                                        const float* __restrict__ bias,
                                                        unsigned short* __restrict__ out,
                                                        long long n) {
    const long long i = (long long)blockIdx.x * 256 + threadIdx.x;
    if (i >= n) return;
    const float v = in[i] + bias[i & 127];
    out[i] = f2bf(fmaxf(v, 0.0f));
}

__global__ __launch_bounds__(256) void k_bias_add(float* __restrict__ out,
                                                  const float* __restrict__ bias,
                                                  long long n) {
    const long long i = (long long)blockIdx.x * 256 + threadIdx.x;
    if (i < n) out[i] += bias[i & 127];
}

// ---------------------------------------------------------------------------
// Launch
// ---------------------------------------------------------------------------
extern "C" void kernel_launch(void* const* d_in, const int* in_sizes, int n_in,
                              void* d_out, int out_size, void* d_ws, size_t ws_size,
                              hipStream_t stream) {
    const float*     input_aux = (const float*)d_in[0];
    const long long* edges     = (const long long*)d_in[1];
    const float*     w_ih      = (const float*)d_in[2];
    const float*     w_hh      = (const float*)d_in[3];
    const float*     b_ih      = (const float*)d_in[4];
    const float*     b_hh      = (const float*)d_in[5];
    const float*     gw1       = (const float*)d_in[6];
    const float*     gb1       = (const float*)d_in[7];
    const float*     gw2       = (const float*)d_in[8];
    const float*     gb2       = (const float*)d_in[9];

    const int       N  = in_sizes[0] / (8 * 128);
    const long long E  = (long long)in_sizes[1] / 2;
    const long long NF = (long long)N * 128;

    char* ws = (char*)d_ws;
    unsigned short* lstmW = (unsigned short*)(ws + OFF_LSTMW);
    unsigned short* g1B   = (unsigned short*)(ws + OFF_G1);
    unsigned short* g2B   = (unsigned short*)(ws + OFF_G2);
    float*          bcomb = (float*)(ws + OFF_BC);
    unsigned short* hnbf  = (unsigned short*)(ws + OFF_HN);   // later reused for relu(x) bf16
    float*          xw    = (float*)(ws + OFF_XW);
    float*          acc1  = (float*)(ws + OFF_ACC);
    float*          outf  = (float*)d_out;

    const int nblk = (N + 127) / 128;

    // --- prep: weight packing (bf16 WMMA fragments) + bias combine ---
    k_pack_lstm<<<1024, 256, 0, stream>>>(w_ih, w_hh, lstmW);
    k_pack_gw<<<64, 256, 0, stream>>>(gw1, g1B);
    k_pack_gw<<<64, 256, 0, stream>>>(gw2, g2B);
    k_bias_comb<<<2, 256, 0, stream>>>(b_ih, b_hh, bcomb);

    // --- LSTM (288 KB dynamic LDS per block) ---
    (void)hipFuncSetAttribute((const void*)k_lstm,
                              hipFuncAttributeMaxDynamicSharedMemorySize, 294912);
    k_lstm<<<nblk, 256, 294912, stream>>>(input_aux, lstmW, bcomb, hnbf, N);

    // --- GCN layer 1: xw = hn @ gw1; acc1 = scatter-add; x = relu(acc1 + gb1) ---
    k_gcn_gemm<<<nblk, 256, 0, stream>>>(hnbf, g1B, xw, N);
    k_zero4<<<(unsigned)((NF / 4 + 255) / 256), 256, 0, stream>>>(acc1, NF / 4);
    k_scatter<<<(unsigned)((E * 32 + 255) / 256), 256, 0, stream>>>(edges, xw, acc1, E);
    k_bias_relu_bf16<<<(unsigned)((NF + 255) / 256), 256, 0, stream>>>(acc1, gb1, hnbf, NF);

    // --- GCN layer 2: xw = x @ gw2; out = scatter-add + gb2 ---
    k_gcn_gemm<<<nblk, 256, 0, stream>>>(hnbf, g2B, xw, N);
    k_zero4<<<(unsigned)((NF / 4 + 255) / 256), 256, 0, stream>>>(outf, NF / 4);
    k_scatter<<<(unsigned)((E * 32 + 255) / 256), 256, 0, stream>>>(edges, xw, outf, E);
    k_bias_add<<<(unsigned)((NF + 255) / 256), 256, 0, stream>>>(outf, gb2, NF);
}
